// GNN_154618823172
// MI455X (gfx1250) — compile-verified, matrix operand order checked by reading, not run on hardware
//
#include <hip/hip_runtime.h>
#include <hip/hip_bf16.h>
#include <stddef.h>

// ---------------------------------------------------------------------------
// 2-layer GCN on gfx1250 / MI455X.
//   layer(x) = A_norm @ (x @ W) + b ,  A_norm = D^-1/2 (A_set + I) D^-1/2
// Dense GEMMs: v_wmma_f32_16x16x32_bf16, operands pre-converted/pre-packed to
// bf16 so the hot loop is pure b128 loads + WMMA. Sparse aggregation via an
// exact-dedup adjacency bit-matrix (atomicOr build, atomic-free gather).
// ---------------------------------------------------------------------------

typedef __attribute__((ext_vector_type(16))) __bf16 v16bf;
typedef __attribute__((ext_vector_type(8)))  __bf16 v8bf;
typedef __attribute__((ext_vector_type(8)))  float  v8f;

union Frag16 { v16bf v; v8bf h[2]; };
union PairU  { __bf16 b[2]; unsigned u; };

#define NODES 8192
#define WORDS 256            // 8192/32 bitmask words per adjacency row

// ---------------------------- WMMA helpers ---------------------------------
__device__ inline v8f wmma_bf16(v16bf a, v16bf b, v8f c) {
  return __builtin_amdgcn_wmma_f32_16x16x32_bf16(false, a, false, b,
                                                 (short)0, c, false, false);
}

// A fragment: bf16 row-major; lane's 16 elems live at p[0..7] and p[16..23]
// (k(r) = 2*(r&3) + 16*(r>>2) + 8*hi, ISA 7.12.2) -> two contiguous b128.
__device__ inline v16bf load_a(const __bf16* __restrict__ p) {
  Frag16 f;
  f.h[0] = *(const v8bf*)(p);
  f.h[1] = *(const v8bf*)(p + 16);
  return f.v;
}

// B fragment: pre-packed; lane's 16 elems are 8 consecutive dwords.
__device__ inline v16bf load_b(const unsigned* __restrict__ p) {
  Frag16 f;
  f.h[0] = *(const v8bf*)(p);
  f.h[1] = *(const v8bf*)(p + 4);
  return f.v;
}

// C/D 16x16 f32: lane L, VGPR r -> (M = r + 8*(L>>4), N = L&15).
__device__ inline void store_tile(float* __restrict__ C, int ldc,
                                  int row_base, int col_base, int lane, v8f acc) {
  const int hi = (lane >> 4) & 1;
  const int n  = col_base + (lane & 15);
#pragma unroll
  for (int r = 0; r < 8; ++r)
    C[(size_t)(row_base + hi * 8 + r) * ldc + n] = acc[r];
}

// ------------------------------- GEMM --------------------------------------
// C[M,N] = A[M,K] @ W[K,N]; A bf16 row-major, W packed bf16, C fp32.
// 256 threads = 8 waves (4Mx2N); block tile 128x128; wave tile 32x64
// (2x4 accumulators -> 8 WMMAs per K-step vs 12 b128-pair fragment loads).
__global__ void __launch_bounds__(256)
wmma_gemm(const __bf16* __restrict__ A, const unsigned* __restrict__ PB,
          float* __restrict__ C, int K, int N) {
  const int lane = threadIdx.x & 31;
  const int wave = threadIdx.x >> 5;       // 0..7
  const int wm = wave >> 1;                // 0..3
  const int wn = wave & 1;                 // 0..1
  const int row0 = blockIdx.y * 128 + wm * 32;
  const int col0 = blockIdx.x * 128 + wn * 64;
  const int hi = (lane >> 4) & 1;
  const int ml = lane & 15;

  const __bf16*  pa = A  + (size_t)(row0 + ml) * K + hi * 8;
  const unsigned* pb = PB + (size_t)(col0 + ml) * 16 + hi * 8;
  const size_t pb_step = (size_t)N * 16;   // dwords per K-tile of packed W

  v8f acc[2][4] = {};

  for (int kt = 0; kt < (K >> 5); ++kt) {
    __builtin_prefetch(pa + 64, 0, 0);               // gfx1250 global_prefetch_b8
    __builtin_prefetch(pb + 2 * pb_step, 0, 0);

    v16bf a0 = load_a(pa);
    v16bf a1 = load_a(pa + (size_t)16 * K);
    v16bf b0 = load_b(pb);
    v16bf b1 = load_b(pb + 256);                     // +16 columns
    v16bf b2 = load_b(pb + 512);
    v16bf b3 = load_b(pb + 768);

    acc[0][0] = wmma_bf16(a0, b0, acc[0][0]);
    acc[0][1] = wmma_bf16(a0, b1, acc[0][1]);
    acc[0][2] = wmma_bf16(a0, b2, acc[0][2]);
    acc[0][3] = wmma_bf16(a0, b3, acc[0][3]);
    acc[1][0] = wmma_bf16(a1, b0, acc[1][0]);
    acc[1][1] = wmma_bf16(a1, b1, acc[1][1]);
    acc[1][2] = wmma_bf16(a1, b2, acc[1][2]);
    acc[1][3] = wmma_bf16(a1, b3, acc[1][3]);

    pa += 32;
    pb += pb_step;
  }

#pragma unroll
  for (int mi = 0; mi < 2; ++mi)
#pragma unroll
    for (int ni = 0; ni < 4; ++ni)
      store_tile(C, N, row0 + mi * 16, col0 + ni * 16, lane, acc[mi][ni]);
}

// -------------------- weight pre-pack (fp32 -> bf16 WMMA order) ------------
// P[(kt*N + n)*16 + hi*8 + r] = {W[k,n], W[k+1,n]} as bf16 pair,
// k = kt*32 + 2*(r&3) + 16*(r>>2) + 8*hi.
__global__ void pack_w_kernel(const float* __restrict__ W,
                              unsigned* __restrict__ P, int K, int N) {
  int idx = blockIdx.x * blockDim.x + threadIdx.x;   // pair index
  if (idx >= (K >> 1) * N) return;
  const int per_tile = N * 16;
  const int kt  = idx / per_tile;
  const int rem = idx - kt * per_tile;
  const int n   = rem >> 4;
  const int q   = rem & 15;
  const int hi  = q >> 3, r = q & 7;
  const int k   = kt * 32 + 2 * (r & 3) + 16 * (r >> 2) + 8 * hi;
  PairU u;
  u.b[0] = (__bf16)W[(size_t)k * N + n];
  u.b[1] = (__bf16)W[(size_t)(k + 1) * N + n];
  P[idx] = u.u;
}

// ---------------------- activations fp32 -> bf16 ---------------------------
__global__ void cvt_bf16_kernel(const float* __restrict__ src,
                                __bf16* __restrict__ dst, int n4) {
  int i = blockIdx.x * blockDim.x + threadIdx.x;
  if (i < n4) {
    const float4 f = ((const float4*)src)[i];
    __bf16* d = dst + (size_t)i * 4;
    d[0] = (__bf16)f.x; d[1] = (__bf16)f.y;
    d[2] = (__bf16)f.z; d[3] = (__bf16)f.w;
  }
}

// ---------------------- adjacency bitmask build ----------------------------
__global__ void edge_mask_kernel(const int* __restrict__ ei,
                                 unsigned* __restrict__ mask, int n_edges) {
  int e = blockIdx.x * blockDim.x + threadIdx.x;
  if (e < n_edges) {
    int r = ei[e];               // edge_index[0, e]
    int c = ei[n_edges + e];     // edge_index[1, e]
    atomicOr(&mask[(size_t)r * WORDS + (c >> 5)], 1u << (c & 31));
  }
}

// degree[i] = popcount(row i) + 1 (eye); dinv = deg^-1/2 (clip implied).
__global__ void degree_kernel(const unsigned* __restrict__ mask,
                              float* __restrict__ dinv) {
  int i = blockIdx.x * blockDim.x + threadIdx.x;
  if (i < NODES) {
    const unsigned* row = mask + (size_t)i * WORDS;
    unsigned s = 0;
#pragma unroll 4
    for (int w = 0; w < WORDS; ++w) s += __popc(row[w]);
    dinv[i] = rsqrtf((float)(s + 1));
  }
}

// ------------------------- sparse aggregation ------------------------------
// out[i,:] = dinv[i]*( sum_{j: bit(i,j)} dinv[j]*Y[j,:] + dinv[i]*Y[i,:] ) + b
// One block per node; uniform bitmask scan (exact dedup, no atomics); each
// thread owns 1-2 feature columns. Optionally ReLU + bf16 output (for the
// next GEMM's A operand).
__global__ void __launch_bounds__(256)
agg_kernel(const float* __restrict__ Y, const unsigned* __restrict__ mask,
           const float* __restrict__ dinv, const float* __restrict__ bias,
           void* __restrict__ outp, int F, int do_relu, int out_bf16) {
  const int i = blockIdx.x;
  const int c0 = threadIdx.x;
  const int c1 = threadIdx.x + 256;
  float acc0 = 0.f, acc1 = 0.f;
  const unsigned* row = mask + (size_t)i * WORDS;

  for (int w = 0; w < WORDS; ++w) {
    unsigned bits = row[w];               // uniform across the block
    while (bits) {
      const int j = (w << 5) + __builtin_ctz(bits);
      bits &= bits - 1;
      const float dj = dinv[j];
      const float* yj = Y + (size_t)j * F;
      acc0 += dj * yj[c0];
      if (F > 256) acc1 += dj * yj[c1];
    }
  }
  const float di = dinv[i];
  const float* yi = Y + (size_t)i * F;

  float r0 = di * (acc0 + di * yi[c0]) + bias[c0];
  if (do_relu) r0 = fmaxf(r0, 0.f);
  if (out_bf16) ((__bf16*)outp)[(size_t)i * F + c0] = (__bf16)r0;
  else          ((float*)outp)[(size_t)i * F + c0] = r0;

  if (F > 256) {
    float r1 = di * (acc1 + di * yi[c1]) + bias[c1];
    if (do_relu) r1 = fmaxf(r1, 0.f);
    if (out_bf16) ((__bf16*)outp)[(size_t)i * F + c1] = (__bf16)r1;
    else          ((float*)outp)[(size_t)i * F + c1] = r1;
  }
}

// ------------------------------ launcher -----------------------------------
extern "C" void kernel_launch(void* const* d_in, const int* in_sizes, int n_in,
                              void* d_out, int out_size, void* d_ws, size_t ws_size,
                              hipStream_t stream) {
  const float* x  = (const float*)d_in[0];   // [8192, 512]
  const int*   ei = (const int*)  d_in[1];   // [2, E]
  const float* W1 = (const float*)d_in[2];   // [512, 512]
  const float* b1 = (const float*)d_in[3];   // [512]
  const float* W2 = (const float*)d_in[4];   // [512, 256]
  const float* b2 = (const float*)d_in[5];   // [256]
  float* out = (float*)d_out;                // [8192, 256]

  const int IN_F = 512, HID = 512, OUT_F = 256;
  const int E = in_sizes[1] / 2;

  // workspace layout (bytes)
  char* ws = (char*)d_ws;
  unsigned* mask = (unsigned*)ws;                               // 8 MB
  size_t off = (size_t)NODES * WORDS * 4;
  float* dinv = (float*)(ws + off);          off += NODES * 4;  // 32 KB
  float* XW   = (float*)(ws + off);          off += (size_t)NODES * HID * 4;   // 16 MB
  __bf16* Xbf = (__bf16*)(ws + off);         off += (size_t)NODES * IN_F * 2;  // 8 MB
  __bf16* Hbf = (__bf16*)(ws + off);         off += (size_t)NODES * HID * 2;   // 8 MB
  unsigned* PW1 = (unsigned*)(ws + off);     off += (size_t)IN_F * HID / 2 * 4; // 512 KB
  unsigned* PW2 = (unsigned*)(ws + off);     // 256 KB

  // graph structure
  hipMemsetAsync(mask, 0, (size_t)NODES * WORDS * 4, stream);
  edge_mask_kernel<<<(E + 255) / 256, 256, 0, stream>>>(ei, mask, E);
  degree_kernel<<<(NODES + 255) / 256, 256, 0, stream>>>(mask, dinv);

  // operand prep
  cvt_bf16_kernel<<<(NODES * IN_F / 4 + 255) / 256, 256, 0, stream>>>(
      x, Xbf, NODES * IN_F / 4);
  pack_w_kernel<<<(IN_F / 2 * HID + 255) / 256, 256, 0, stream>>>(W1, PW1, IN_F, HID);
  pack_w_kernel<<<(HID / 2 * OUT_F + 255) / 256, 256, 0, stream>>>(W2, PW2, HID, OUT_F);

  // layer 1: XW = x @ W1 ; Hbf = bf16(relu(A_norm @ XW + b1))
  dim3 g1(HID / 128, NODES / 128);
  wmma_gemm<<<g1, 256, 0, stream>>>(Xbf, PW1, XW, IN_F, HID);
  agg_kernel<<<NODES, 256, 0, stream>>>(XW, mask, dinv, b1, Hbf, HID, 1, 1);

  // layer 2: XW2 = H @ W2 ; out = A_norm @ XW2 + b2
  dim3 g2(OUT_F / 128, NODES / 128);
  wmma_gemm<<<g2, 256, 0, stream>>>(Hbf, PW2, XW, HID, OUT_F);
  agg_kernel<<<NODES, 256, 0, stream>>>(XW, mask, dinv, b2, out, OUT_F, 0, 0);
}
